// MPSContextEntangler_82497731821675
// MI455X (gfx1250) — matrix operand order repaired and checked by reading, not compiled, not run on hardware
//
#include <hip/hip_runtime.h>
#include <stddef.h>
#include <stdint.h>

// ---------------------------------------------------------------------------
// MPSContextEntangler for MI455X (gfx1250, wave32)
//
//   K0a: emb fp32 -> bf16
//   K0b: W  fp32 -> W^T bf16 (so B fragments gather contiguous-K from LDS)
//   K1 : site(bf16) = emb @ W + b   via v_wmma_f32_16x16x32_bf16,
//        double-buffered LDS staging via GLOBAL_LOAD_ASYNC_TO_LDS (ASYNCcnt)
//   K2 : per (batch, chunk-of-16): product of 32x32 transfer matrices
//   K3 : Kogge-Stone inclusive scan of chunk matrices (8 rounds)
//   K4 : per chunk: reconstruct L from prefix, replay 16 steps with async
//        LDS tile prefetch: Tm, gram, 4x4 Jacobi -> entropy, gate, L update
//   K5 : context = emb * gate
//   K6 : entropy = ents[:, :-1]
// ---------------------------------------------------------------------------

typedef __attribute__((ext_vector_type(16))) __bf16        v16bf;
typedef __attribute__((ext_vector_type(8)))  float         v8f;
typedef __attribute__((ext_vector_type(4)))  float         v4f;
typedef __attribute__((ext_vector_type(4)))  unsigned int  v4u;
typedef __attribute__((ext_vector_type(4)))  int           v4i;

union FragU { v4u q[2]; v16bf h; };

// ------------------------- gfx1250 async-to-LDS support (builtin-guarded)
#if defined(__has_builtin)
#if __has_builtin(__builtin_amdgcn_global_load_async_to_lds_b128)
#define HAVE_ASYNC 1
#endif
#endif
#ifndef HAVE_ASYNC
#define HAVE_ASYNC 0
#endif

#if HAVE_ASYNC
typedef __attribute__((address_space(1))) v4i as1_v4i;  // global <4 x i32>
typedef __attribute__((address_space(3))) v4i as3_v4i;  // LDS    <4 x i32>
__device__ __forceinline__ void async_b128(const void* g, void* l) {
  // per-lane 16B global->LDS copy, tracked by ASYNCcnt
  __builtin_amdgcn_global_load_async_to_lds_b128(
      (as1_v4i*)(uintptr_t)g,
      (as3_v4i*)(unsigned int)(uintptr_t)l,  // low 32 bits of flat LDS addr = LDS offset
      0, 0);
}
#if __has_builtin(__builtin_amdgcn_s_wait_asynccnt)
#define WAIT_ASYNCCNT(n) __builtin_amdgcn_s_wait_asynccnt(n)
#else
#define WAIT_ASYNCCNT(n) asm volatile("s_wait_asynccnt %0" ::"i"(n) : "memory")
#endif
#endif

__device__ __forceinline__ unsigned short f32_to_bf16(float f) {
  unsigned int u = __float_as_uint(f);
  u += 0x7FFFu + ((u >> 16) & 1u);  // round to nearest even
  return (unsigned short)(u >> 16);
}
__device__ __forceinline__ float bf16_to_f32(unsigned short u) {
  return __uint_as_float(((unsigned int)u) << 16);
}

// ---------------------------------------------------------------- K0a: f32->bf16
__global__ __launch_bounds__(256) void k0_f2bf(const float* __restrict__ src,
                                               unsigned short* __restrict__ dst,
                                               int n) {
  int i = blockIdx.x * 256 + threadIdx.x;
  const int stride = gridDim.x * 256;
  for (; i < n; i += stride) dst[i] = f32_to_bf16(src[i]);
}

// ---------------------------------------------------------------- K0b: W -> W^T (bf16)
__global__ __launch_bounds__(256) void k0_transposeW(const float* __restrict__ W,
                                                     unsigned short* __restrict__ Wt) {
  __shared__ float tile[32][33];
  const int k0 = blockIdx.y * 32, n0 = blockIdx.x * 32;
  const int c = threadIdx.x & 31, r0 = threadIdx.x >> 5;
#pragma unroll
  for (int it = 0; it < 4; ++it) {
    const int r = r0 + it * 8;
    tile[r][c] = W[(size_t)(k0 + r) * 4096 + n0 + c];
  }
  __syncthreads();
#pragma unroll
  for (int it = 0; it < 4; ++it) {
    const int r = r0 + it * 8;
    Wt[(size_t)(n0 + r) * 1024 + k0 + c] = f32_to_bf16(tile[c][r]);
  }
}

// ---------------------------------------------------------------- K1: WMMA GEMM
// A fragment (16x32 bf16): lane m (0-15) holds K {0..7,16..23}, lane m+16 holds
// K {8..15,24..31}. B fragment (32x16): same K split per lane half, col n.
__device__ __forceinline__ v16bf frag_a(const unsigned short* S, int ld, int mbase,
                                        int ks, int lane) {
  const int m  = mbase + (lane & 15);
  const int kb = (lane < 16) ? 0 : 8;
  const unsigned short* p = S + m * ld + ks + kb;
  FragU u;
  u.q[0] = *(const v4u*)(p);
  u.q[1] = *(const v4u*)(p + 16);
  return u.h;
}
__device__ __forceinline__ v16bf frag_b(const unsigned short* S, int ld, int nbase,
                                        int ks, int lane) {
  const int n  = nbase + (lane & 15);
  const int kb = (lane < 16) ? 0 : 8;
  const unsigned short* p = S + n * ld + ks + kb;
  FragU u;
  u.q[0] = *(const v4u*)(p);
  u.q[1] = *(const v4u*)(p + 16);
  return u.h;
}

#if HAVE_ASYNC
// Async double-buffered variant: block tile 128x128, K-step 64, 16 iterations.
#define LDK 72  // 64 + pad; rows stay 16B aligned (144B)
__global__ __launch_bounds__(256) void k1_gemm(const unsigned short* __restrict__ Abf,
                                               const unsigned short* __restrict__ Wt,
                                               const float* __restrict__ bias,
                                               unsigned short* __restrict__ site) {
  __shared__ unsigned short As[2][128 * LDK];
  __shared__ unsigned short Bs[2][128 * LDK];

  const int tid   = threadIdx.x;
  const int lane  = tid & 31;
  const int wid   = tid >> 5;
  const int m0    = blockIdx.y * 128;
  const int n0    = blockIdx.x * 128;
  const int m_off = (wid & 3) * 32;
  const int n_off = (wid >> 2) * 64;

  const v8f vz = {0.f, 0.f, 0.f, 0.f, 0.f, 0.f, 0.f, 0.f};
  v8f acc[2][4];
#pragma unroll
  for (int i = 0; i < 2; ++i)
#pragma unroll
    for (int c = 0; c < 4; ++c) acc[i][c] = vz;

  const int srow = tid >> 1;  // 0..127
  const int half = tid & 1;   // which 64B half of the 128B-wide K slab
  const unsigned short* gA = Abf + (size_t)(m0 + srow) * 1024 + half * 32;
  const unsigned short* gB = Wt  + (size_t)(n0 + srow) * 1024 + half * 32;
  unsigned short* lA[2] = {&As[0][srow * LDK + half * 32], &As[1][srow * LDK + half * 32]};
  unsigned short* lB[2] = {&Bs[0][srow * LDK + half * 32], &Bs[1][srow * LDK + half * 32]};

  auto issue = [&](int kk, int buf) {  // 8 async b128 per thread = one 128x64 A+B tile
#pragma unroll
    for (int j = 0; j < 4; ++j) {
      async_b128(gA + kk + j * 8, lA[buf] + j * 8);
      async_b128(gB + kk + j * 8, lB[buf] + j * 8);
    }
  };

  issue(0, 0);
  int buf = 0;
  for (int it = 0; it < 16; ++it) {
    if (it < 15) {
      issue((it + 1) * 64, buf ^ 1);  // fetch next tile into other buffer
      WAIT_ASYNCCNT(8);               // previous 8 (current tile) complete in order
    } else {
      WAIT_ASYNCCNT(0);
    }
    __syncthreads();

    const unsigned short* Ab = As[buf];
    const unsigned short* Bb = Bs[buf];
#pragma unroll
    for (int ks = 0; ks < 64; ks += 32) {
      v16bf af[2], bfr[4];
      af[0] = frag_a(Ab, LDK, m_off, ks, lane);
      af[1] = frag_a(Ab, LDK, m_off + 16, ks, lane);
#pragma unroll
      for (int c = 0; c < 4; ++c) bfr[c] = frag_b(Bb, LDK, n_off + c * 16, ks, lane);
#pragma unroll
      for (int i = 0; i < 2; ++i)
#pragma unroll
        for (int c = 0; c < 4; ++c)
          acc[i][c] = __builtin_amdgcn_wmma_f32_16x16x32_bf16(
              false, af[i], false, bfr[c], (short)0, acc[i][c], false, false);
    }
    __syncthreads();
    buf ^= 1;
  }

  const int mrow_add = (lane < 16) ? 0 : 8;
  const int ncol     = lane & 15;
  float bv[4];
#pragma unroll
  for (int c = 0; c < 4; ++c) bv[c] = bias[n0 + n_off + c * 16 + ncol];
#pragma unroll
  for (int i = 0; i < 2; ++i)
#pragma unroll
    for (int r = 0; r < 8; ++r) {
      const size_t trow = (size_t)(m0 + m_off + i * 16 + r + mrow_add);
#pragma unroll
      for (int c = 0; c < 4; ++c) {
        const int nl = n_off + c * 16 + ncol;
        site[trow * 4096 + n0 + nl] = f32_to_bf16(acc[i][c][r] + bv[c]);
      }
    }
}
#else   // ------------- synchronous fallback (K-step 32)
__global__ __launch_bounds__(256) void k1_gemm(const unsigned short* __restrict__ Abf,
                                               const unsigned short* __restrict__ Wt,
                                               const float* __restrict__ bias,
                                               unsigned short* __restrict__ site) {
  __shared__ unsigned short As[128 * 40];
  __shared__ unsigned short Bs[128 * 40];

  const int tid   = threadIdx.x;
  const int lane  = tid & 31;
  const int wid   = tid >> 5;
  const int m0    = blockIdx.y * 128;
  const int n0    = blockIdx.x * 128;
  const int m_off = (wid & 3) * 32;
  const int n_off = (wid >> 2) * 64;

  const v8f vz = {0.f, 0.f, 0.f, 0.f, 0.f, 0.f, 0.f, 0.f};
  v8f acc[2][4];
#pragma unroll
  for (int i = 0; i < 2; ++i)
#pragma unroll
    for (int c = 0; c < 4; ++c) acc[i][c] = vz;

  const int srow = tid >> 1;
  const int half = tid & 1;

  for (int kk = 0; kk < 1024; kk += 32) {
    {
      const v4u* g = (const v4u*)(Abf + (size_t)(m0 + srow) * 1024 + kk + half * 16);
      v4u* d = (v4u*)(As + srow * 40 + half * 16);
      d[0] = g[0]; d[1] = g[1];
    }
    {
      const v4u* g = (const v4u*)(Wt + (size_t)(n0 + srow) * 1024 + kk + half * 16);
      v4u* d = (v4u*)(Bs + srow * 40 + half * 16);
      d[0] = g[0]; d[1] = g[1];
    }
    __syncthreads();

    v16bf af[2], bfr[4];
    af[0] = frag_a(As, 40, m_off, 0, lane);
    af[1] = frag_a(As, 40, m_off + 16, 0, lane);
#pragma unroll
    for (int c = 0; c < 4; ++c) bfr[c] = frag_b(Bs, 40, n_off + c * 16, 0, lane);
#pragma unroll
    for (int i = 0; i < 2; ++i)
#pragma unroll
      for (int c = 0; c < 4; ++c)
        acc[i][c] = __builtin_amdgcn_wmma_f32_16x16x32_bf16(
            false, af[i], false, bfr[c], (short)0, acc[i][c], false, false);
    __syncthreads();
  }

  const int mrow_add = (lane < 16) ? 0 : 8;
  const int ncol     = lane & 15;
  float bv[4];
#pragma unroll
  for (int c = 0; c < 4; ++c) bv[c] = bias[n0 + n_off + c * 16 + ncol];
#pragma unroll
  for (int i = 0; i < 2; ++i)
#pragma unroll
    for (int r = 0; r < 8; ++r) {
      const size_t trow = (size_t)(m0 + m_off + i * 16 + r + mrow_add);
#pragma unroll
      for (int c = 0; c < 4; ++c) {
        const int nl = n_off + c * 16 + ncol;
        site[trow * 4096 + n0 + nl] = f32_to_bf16(acc[i][c][r] + bv[c]);
      }
    }
}
#endif

// ---------------------------------------------------------------- K2: chunk products
__global__ __launch_bounds__(128) void k2_chunk(const unsigned short* __restrict__ site,
                                                float* __restrict__ Cout) {
  __shared__ float Csm[1024], Msm[1024], red[128];
  const int tid = threadIdx.x;
  const int b = blockIdx.y, k = blockIdx.x;

#pragma unroll
  for (int q = 0; q < 8; ++q) {
    const int e = tid * 8 + q;
    Csm[e] = ((e >> 5) == (e & 31)) ? 1.f : 0.f;
  }
  __syncthreads();

  const int row = tid >> 2, col0 = (tid & 3) * 8;

  for (int tt = 0; tt < 16; ++tt) {
    const unsigned short* tp = site + ((size_t)b * 4096 + (size_t)k * 16 + tt) * 4096;
#pragma unroll
    for (int q = 0; q < 8; ++q) {  // M_t[i][j] = sum_s site[t, i*128+s*32+j]
      const int e = tid * 8 + q, i = e >> 5, j = e & 31;
      Msm[e] = bf16_to_f32(tp[i * 128 + j])      + bf16_to_f32(tp[i * 128 + 32 + j]) +
               bf16_to_f32(tp[i * 128 + 64 + j]) + bf16_to_f32(tp[i * 128 + 96 + j]);
    }
    __syncthreads();

    float s[8];
#pragma unroll
    for (int q = 0; q < 8; ++q) s[q] = 0.f;
    for (int c = 0; c < 32; ++c) {
      const float cv = Csm[row * 32 + c];
      const float* mrow = Msm + c * 32 + col0;
#pragma unroll
      for (int q = 0; q < 8; ++q) s[q] += cv * mrow[q];
    }
    float loc = 0.f;
#pragma unroll
    for (int q = 0; q < 8; ++q) loc += s[q] * s[q];
    red[tid] = loc;
    __syncthreads();
    for (int off = 64; off > 0; off >>= 1) {
      if (tid < off) red[tid] += red[tid + off];
      __syncthreads();
    }
    const float rn = rsqrtf(red[0] + 1e-30f);
    __syncthreads();
#pragma unroll
    for (int q = 0; q < 8; ++q) Csm[row * 32 + col0 + q] = s[q] * rn;
    __syncthreads();
  }

  float* Co = Cout + ((size_t)b * 256 + k) * 1024;
#pragma unroll
  for (int q = 0; q < 8; ++q) Co[tid * 8 + q] = Csm[tid * 8 + q];
}

// ---------------------------------------------------------------- K3: scan round
__global__ __launch_bounds__(128) void k3_scan(const float* __restrict__ src,
                                               float* __restrict__ dst, int stridek) {
  __shared__ float Asm[1024], Bsm[1024], red[128];
  const int tid = threadIdx.x;
  const int b = blockIdx.y, k = blockIdx.x;
  const float* Bp = src + ((size_t)b * 256 + k) * 1024;
  float* Dp = dst + ((size_t)b * 256 + k) * 1024;

  if (k < stridek) {
#pragma unroll
    for (int q = 0; q < 8; ++q) Dp[tid * 8 + q] = Bp[tid * 8 + q];
    return;
  }
  const float* Ap = src + ((size_t)b * 256 + k - stridek) * 1024;
#pragma unroll
  for (int q = 0; q < 8; ++q) {
    Asm[tid + 128 * q] = Ap[tid + 128 * q];
    Bsm[tid + 128 * q] = Bp[tid + 128 * q];
  }
  __syncthreads();

  const int row = tid >> 2, col0 = (tid & 3) * 8;
  float s[8];
#pragma unroll
  for (int q = 0; q < 8; ++q) s[q] = 0.f;
  for (int c = 0; c < 32; ++c) {
    const float av = Asm[row * 32 + c];
    const float* brow = Bsm + c * 32 + col0;
#pragma unroll
    for (int q = 0; q < 8; ++q) s[q] += av * brow[q];
  }
  float loc = 0.f;
#pragma unroll
  for (int q = 0; q < 8; ++q) loc += s[q] * s[q];
  red[tid] = loc;
  __syncthreads();
  for (int off = 64; off > 0; off >>= 1) {
    if (tid < off) red[tid] += red[tid + off];
    __syncthreads();
  }
  const float rn = rsqrtf(red[0] + 1e-30f);
#pragma unroll
  for (int q = 0; q < 8; ++q) Dp[row * 32 + col0 + q] = s[q] * rn;
}

// ---------------------------------------------------------------- K4: within-chunk replay
__device__ __forceinline__ void eig4(float G[4][4], float lam[4]) {
  for (int sweep = 0; sweep < 8; ++sweep)
    for (int p = 0; p < 3; ++p)
      for (int q = p + 1; q < 4; ++q) {
        const float apq = G[p][q];
        if (fabsf(apq) < 1e-20f) continue;
        const float tau = (G[q][q] - G[p][p]) / (2.f * apq);
        const float t   = ((tau >= 0.f) ? 1.f : -1.f) / (fabsf(tau) + sqrtf(1.f + tau * tau));
        const float c   = 1.f / sqrtf(1.f + t * t);
        const float s   = t * c;
        for (int r = 0; r < 4; ++r) {
          const float grp = G[r][p], grq = G[r][q];
          G[r][p] = c * grp - s * grq;
          G[r][q] = s * grp + c * grq;
        }
        for (int r = 0; r < 4; ++r) {
          const float gpr = G[p][r], gqr = G[q][r];
          G[p][r] = c * gpr - s * gqr;
          G[q][r] = s * gpr + c * gqr;
        }
      }
  for (int i = 0; i < 4; ++i) lam[i] = G[i][i];
}

__global__ __launch_bounds__(128) void k4_walk(const unsigned short* __restrict__ site,
                                               const float* __restrict__ Sincl,
                                               float* __restrict__ gates,
                                               float* __restrict__ ents) {
  __shared__ float L[32], tm[128], red[128], gsm[16];
#if HAVE_ASYNC
  __shared__ unsigned short tbuf[2][4096];
#endif
  const int tid = threadIdx.x;
  const int b = blockIdx.y, k = blockIdx.x;

  if (tid < 32) {  // L at chunk start: dir(L0^T * prefix_{k-1}); L0 = ones/sqrt(32)
    if (k == 0) {
      L[tid] = 0.17677669529663688f;
    } else {
      const float* P = Sincl + ((size_t)b * 256 + (k - 1)) * 1024;
      float s = 0.f;
      for (int i = 0; i < 32; ++i) s += P[i * 32 + tid];
      float v2 = s * s;
#pragma unroll
      for (int off = 16; off > 0; off >>= 1) v2 += __shfl_xor(v2, off, 32);
      L[tid] = s / (sqrtf(v2) + 1e-6f);
    }
  }

#if HAVE_ASYNC
  {
    const unsigned short* tp0 = site + ((size_t)b * 4096 + (size_t)k * 16) * 4096;
#pragma unroll
    for (int j = 0; j < 4; ++j)
      async_b128(tp0 + tid * 32 + j * 8, &tbuf[0][tid * 32 + j * 8]);
  }
#endif
  __syncthreads();

  for (int tt = 0; tt < 16; ++tt) {
    const size_t t = (size_t)k * 16 + tt;
    float a = 0.f;
#if HAVE_ASYNC
    const int cur = tt & 1;
    if (tt < 15) {  // prefetch next tile into other buffer while this one computes
      const unsigned short* tpn = site + ((size_t)b * 4096 + t + 1) * 4096;
#pragma unroll
      for (int j = 0; j < 4; ++j)
        async_b128(tpn + tid * 32 + j * 8, &tbuf[cur ^ 1][tid * 32 + j * 8]);
      WAIT_ASYNCCNT(4);
    } else {
      WAIT_ASYNCCNT(0);
    }
    __syncthreads();
    for (int i = 0; i < 32; ++i) a += L[i] * bf16_to_f32(tbuf[cur][i * 128 + tid]);
#else
    const unsigned short* tp = site + ((size_t)b * 4096 + t) * 4096;
    if (tt < 15) __builtin_prefetch(tp + 4096 + tid * 32, 0, 1);
    for (int i = 0; i < 32; ++i) a += L[i] * bf16_to_f32(tp[i * 128 + tid]);
#endif
    tm[tid] = a;
    red[tid] = a * a;
    __syncthreads();
    for (int off = 64; off > 0; off >>= 1) {
      if (tid < off) red[tid] += red[tid + off];
      __syncthreads();
    }
    const float fro = red[0];

    if (tid < 32) {  // v = sum_s Tm[s,:], then L <- v/(||v||+1e-6)
      float v = tm[tid] + tm[32 + tid] + tm[64 + tid] + tm[96 + tid];
      float v2 = v * v;
#pragma unroll
      for (int off = 16; off > 0; off >>= 1) v2 += __shfl_xor(v2, off, 32);
      L[tid] = v / (sqrtf(v2) + 1e-6f);
    }
    if (tid >= 32 && tid < 48) {  // gram = Tm Tm^T
      const int s1 = (tid - 32) >> 2, s2 = (tid - 32) & 3;
      float g = 0.f;
      for (int j = 0; j < 32; ++j) g += tm[s1 * 32 + j] * tm[s2 * 32 + j];
      gsm[tid - 32] = g;
    }
    __syncthreads();

    if (tid == 0) {
      float G[4][4], lam[4];
      for (int s1 = 0; s1 < 4; ++s1)
        for (int s2 = 0; s2 < 4; ++s2) G[s1][s2] = gsm[s1 * 4 + s2];
      eig4(G, lam);
      float pl[4], sum = 0.f;
#pragma unroll
      for (int i = 0; i < 4; ++i) { pl[i] = fmaxf(lam[i], 1e-12f); sum += pl[i]; }
      float ent = 0.f;
#pragma unroll
      for (int i = 0; i < 4; ++i) { const float p = pl[i] / sum; ent -= p * logf(p); }
      gates[(size_t)b * 4096 + t] = tanhf(sqrtf(fro + 1e-12f));
      ents[(size_t)b * 4096 + t]  = ent;
    }
    __syncthreads();
  }
}

// ---------------------------------------------------------------- K5: context = emb * gate
__global__ __launch_bounds__(256) void k5_ctx(const float* __restrict__ emb,
                                              const float* __restrict__ gates,
                                              float* __restrict__ out) {
  const int i = blockIdx.x * 256 + threadIdx.x;
  const v4f e = ((const v4f*)emb)[i];
  const float g = gates[i >> 8];
  ((v4f*)out)[i] = e * g;
}

// ---------------------------------------------------------------- K6: entropy[:, :-1]
__global__ __launch_bounds__(256) void k6_ent(const float* __restrict__ ents,
                                              float* __restrict__ out, int n) {
  const int i = blockIdx.x * 256 + threadIdx.x;
  if (i < n) {
    const int bb = i / 4095;
    const int t  = i - bb * 4095;
    out[i] = ents[bb * 4096 + t];
  }
}

// ---------------------------------------------------------------- launcher
extern "C" void kernel_launch(void* const* d_in, const int* in_sizes, int n_in,
                              void* d_out, int out_size, void* d_ws, size_t ws_size,
                              hipStream_t stream) {
  (void)in_sizes; (void)n_in; (void)out_size; (void)ws_size;

  const float* emb  = (const float*)d_in[0];
  const float* W    = (const float*)d_in[1];
  const float* bias = (const float*)d_in[2];
  float* out = (float*)d_out;

  char* ws = (char*)d_ws;
  const size_t off_site  = 0;
  const size_t off_embbf = off_site  + 268435456ULL;
  const size_t off_wt    = off_embbf + 67108864ULL;
  const size_t off_ca    = off_wt    + 8388608ULL;
  const size_t off_cb    = off_ca    + 8388608ULL;
  const size_t off_g     = off_cb    + 8388608ULL;
  const size_t off_e     = off_g     + 131072ULL;

  unsigned short* site  = (unsigned short*)(ws + off_site);
  unsigned short* embbf = (unsigned short*)(ws + off_embbf);
  unsigned short* wt    = (unsigned short*)(ws + off_wt);
  float* CA    = (float*)(ws + off_ca);
  float* CB    = (float*)(ws + off_cb);
  float* gates = (float*)(ws + off_g);
  float* entsw = (float*)(ws + off_e);

  k0_f2bf<<<8192, 256, 0, stream>>>(emb, embbf, 33554432);
  k0_transposeW<<<dim3(128, 32), 256, 0, stream>>>(W, wt);
  k1_gemm<<<dim3(32, 256), 256, 0, stream>>>(embbf, wt, bias, site);
  k2_chunk<<<dim3(256, 8), 128, 0, stream>>>(site, CA);

  float* a = CA;
  float* b = CB;
  for (int r = 0; r < 8; ++r) {  // even # of rounds -> inclusive scan ends in CA
    k3_scan<<<dim3(256, 8), 128, 0, stream>>>(a, b, 1 << r);
    float* tpp = a; a = b; b = tpp;
  }

  k4_walk<<<dim3(256, 8), 128, 0, stream>>>(site, a, gates, entsw);
  k5_ctx<<<32768, 256, 0, stream>>>(emb, gates, out);
  k6_ent<<<128, 256, 0, stream>>>(entsw, out + 33554432, 8 * 4095);
}
